// TGN_34230889349755
// MI455X (gfx1250) — matrix operand order repaired and checked by reading, not compiled, not run on hardware
//
#include <hip/hip_runtime.h>
#include <math.h>

typedef __attribute__((ext_vector_type(16))) _Float16 v16h;
typedef __attribute__((ext_vector_type(8)))  _Float16 v8h;
typedef __attribute__((ext_vector_type(8)))  float    v8f;
typedef _Float16 h16;

#define NDN 200000   // nodes
#define NBB 300      // batch
#define KNB 10       // neighbors
#define DD  172      // feature dim
#define DP  192      // 172 padded (mult of 32, usable as both N-pad and K-pad)
#define QDIM 344
#define QP  352      // 344 padded (22*16, 11*32)
#define KDIM 516
#define KP  544      // 516 padded (17*32)
#define RAWD 689
#define RAWH 344
#define MSGD 100

// ---------------------------------------------------------------- weight prep
__global__ void k_conv_weight(const float* __restrict__ W, int Kd, int Nd,
                              h16* __restrict__ Bt, int Kp, int Np) {
  int idx = blockIdx.x * blockDim.x + threadIdx.x;
  if (idx >= Kp * Np) return;
  int n = idx / Kp, k = idx % Kp;
  float v = (n < Nd && k < Kd) ? W[k * Nd + n] : 0.0f;
  Bt[idx] = (h16)v;
}

__global__ void k_pad_bias(const float* __restrict__ b, int Nd,
                           float* __restrict__ o, int Np) {
  int t = blockIdx.x * blockDim.x + threadIdx.x;
  if (t < Np) o[t] = (t < Nd) ? b[t] : 0.0f;
}

// ---------------------------------------------------------------- memory delta
__global__ void k_init_map(int* __restrict__ map) {
  int t = blockIdx.x * blockDim.x + threadIdx.x;
  if (t < NDN) map[t] = -1;
}

__global__ void k_scatter_map(const int* __restrict__ src, const int* __restrict__ dst,
                              int* __restrict__ map) {
  int t = blockIdx.x * blockDim.x + threadIdx.x;
  if (t < NBB) map[src[t]] = t;
  else if (t < 2 * NBB) map[dst[t - NBB]] = t;
}

__global__ void k_nodes(const int* __restrict__ src, const int* __restrict__ dst,
                        const int* __restrict__ neg, int* __restrict__ nodes) {
  int t = blockIdx.x * blockDim.x + threadIdx.x;
  if (t >= 3 * NBB) return;
  nodes[t] = (t < NBB) ? src[t] : (t < 2 * NBB) ? dst[t - NBB] : neg[t - 2 * NBB];
}

// message MLP + GRU memory update for the 2B touched rows (reads ORIGINAL memory)
__global__ void k_msggru(const float* __restrict__ memf, const float* __restrict__ edgef,
                         const float* __restrict__ last_update,
                         const float* __restrict__ time_w, const float* __restrict__ time_b,
                         const float* __restrict__ num_w1, const float* __restrict__ num_b1,
                         const float* __restrict__ num_w2, const float* __restrict__ num_b2,
                         const float* __restrict__ msg_w1, const float* __restrict__ msg_b1,
                         const float* __restrict__ msg_w2, const float* __restrict__ msg_b2,
                         const float* __restrict__ gru_wih, const float* __restrict__ gru_whh,
                         const float* __restrict__ gru_bih, const float* __restrict__ gru_bhh,
                         const float* __restrict__ num_conn, const float* __restrict__ edge_times,
                         const int* __restrict__ src, const int* __restrict__ dst,
                         const int* __restrict__ edge_idxs, float* __restrict__ upd) {
  __shared__ float raw[RAWD];
  __shared__ float h1s[RAWH];
  __shared__ float msg[MSGD];
  __shared__ float gis[3 * DD];
  __shared__ float ghs[3 * DD];
  int r = blockIdx.x;
  int i = (r < NBB) ? r : r - NBB;
  int a = (r < NBB) ? src[i] : dst[i];
  int b = (r < NBB) ? dst[i] : src[i];
  int t = threadIdx.x;
  float ne = num_conn[i] * num_w1[0] + num_b1[0];
  ne = fmaxf(ne, 0.0f) * num_w2[0] + num_b2[0];
  float dt = edge_times[i] - last_update[a];
  for (int d = t; d < RAWD; d += blockDim.x) {
    float v;
    if      (d <     DD) v = memf[a * DD + d];
    else if (d < 2 * DD) v = memf[b * DD + d - DD];
    else if (d < 3 * DD) v = edgef[edge_idxs[i] * DD + d - 2 * DD];
    else if (d < 4 * DD) v = cosf(dt * time_w[d - 3 * DD] + time_b[d - 3 * DD]);
    else v = ne;
    raw[d] = v;
  }
  __syncthreads();
  for (int j = t; j < RAWH; j += blockDim.x) {
    float acc = msg_b1[j];
    for (int k = 0; k < RAWD; ++k) acc += raw[k] * msg_w1[k * RAWH + j];
    h1s[j] = fmaxf(acc, 0.0f);
  }
  __syncthreads();
  for (int j = t; j < MSGD; j += blockDim.x) {
    float acc = msg_b2[j];
    for (int k = 0; k < RAWH; ++k) acc += h1s[k] * msg_w2[k * MSGD + j];
    msg[j] = acc;
  }
  __syncthreads();
  for (int j = t; j < 3 * DD; j += blockDim.x) {
    float ai = gru_bih[j], ah = gru_bhh[j];
    for (int k = 0; k < MSGD; ++k) ai += msg[k] * gru_wih[j * MSGD + k];
    for (int k = 0; k < DD;   ++k) ah += memf[a * DD + k] * gru_whh[j * DD + k];
    gis[j] = ai; ghs[j] = ah;
  }
  __syncthreads();
  for (int d = t; d < DD; d += blockDim.x) {
    float rg = 1.0f / (1.0f + expf(-(gis[d] + ghs[d])));
    float z  = 1.0f / (1.0f + expf(-(gis[DD + d] + ghs[DD + d])));
    float nn = tanhf(gis[2 * DD + d] + rg * ghs[2 * DD + d]);
    upd[r * DD + d] = (1.0f - z) * nn + z * memf[a * DD + d];
  }
}

// feat(n) = (updated?)memory[n] + node_features[n]  -> f16 [Mpad][DP]
__global__ void k_gather_feat(const int* __restrict__ idx, int Mact, int Mpad,
                              const int* __restrict__ map, const float* __restrict__ upd,
                              const float* __restrict__ memf, const float* __restrict__ nodef,
                              h16* __restrict__ out) {
  int g = blockIdx.x * blockDim.x + threadIdx.x;
  if (g >= Mpad * DP) return;
  int i = g / DP, d = g % DP;
  float v = 0.0f;
  if (i < Mact && d < DD) {
    int n = idx[i];
    int mi = map[n];
    float m = (mi >= 0) ? upd[mi * DD + d] : memf[n * DD + d];
    v = m + nodef[n * DD + d];
  }
  out[g] = (h16)v;
}

// ---------------------------------------------------------------- attention inputs
__global__ void k_build_qin(const h16* __restrict__ srcb, const float* __restrict__ time_b,
                            h16* __restrict__ qin, int Mact, int Mpad) {
  int g = blockIdx.x * blockDim.x + threadIdx.x;
  if (g >= Mpad * QP) return;
  int i = g / QP, c = g % QP;
  float v = 0.0f;
  if (i < Mact) {
    if (c < DD) v = (float)srcb[i * DP + c];
    else if (c < QDIM) v = cosf(time_b[c - DD]);   // time_enc(0)
  }
  qin[g] = (h16)v;
}

__global__ void k_build_kin(int mode, const int* __restrict__ nb, const int* __restrict__ eidx,
                            const float* __restrict__ et, const h16* __restrict__ neighb,
                            const int* __restrict__ map, const float* __restrict__ upd,
                            const float* __restrict__ memf, const float* __restrict__ nodef,
                            const float* __restrict__ edgef, const float* __restrict__ edge_times,
                            const float* __restrict__ time_w, const float* __restrict__ time_b,
                            int rep, int seg0, int Mact,
                            h16* __restrict__ kin, int Ppad) {
  int g = blockIdx.x * blockDim.x + threadIdx.x;
  if (g >= Ppad * KP) return;
  int p = g / KP, c = g % KP;
  int i = p / KNB, kk = p % KNB;
  float v = 0.0f;
  if (i < Mact) {
    int gr = seg0 + i;
    int gp = gr * KNB + kk;
    if (c < DD) {
      if (mode == 0) {
        int n = nb[gp];
        int mi = map[n];
        float m = (mi >= 0) ? upd[mi * DD + c] : memf[n * DD + c];
        v = m + nodef[n * DD + c];
      } else {
        v = (float)neighb[(size_t)gp * DP + c];
      }
    } else if (c < 2 * DD) {
      v = edgef[eidx[gp] * DD + (c - DD)];
    } else if (c < 3 * DD) {
      int d = c - 2 * DD;
      float tsv = edge_times[(gr / rep) % NBB];
      v = cosf((tsv - et[gp]) * time_w[d] + time_b[d]);
    }
  }
  kin[g] = (h16)v;
}

// ---------------------------------------------------------------- WMMA GEMM
// C[M,Np] = A[M,Kp](f16,row-major) x Bt[Np,Kp](f16 = W^T, row-major) + bias
// A- and B-fragments both in contiguous K-split-lane-halves order (element s = k 16*hi + s).
__global__ void k_gemm(const h16* __restrict__ A, int lda,
                       const h16* __restrict__ Bt, int Kp,
                       const float* __restrict__ bias, int relu,
                       h16* __restrict__ C16, float* __restrict__ C32, int ldc,
                       int Np, int Mact, int Nact) {
  int wave = threadIdx.x >> 5;
  int lane = threadIdx.x & 31;
  int n0 = (blockIdx.y * 8 + wave) * 16;
  if (n0 >= Np) return;
  int m0 = blockIdx.x * 16;
  int hi  = lane >> 4;          // 0 or 1
  int akb = hi ? 16 : 0;        // A: lanes 16-31 hold K=16..31
  int bkb = hi ? 16 : 0;        // B: lanes 16-31 hold K=16..31
  const h16* ap = A  + (size_t)(m0 + (lane & 15)) * lda + akb;
  const h16* bp = Bt + (size_t)(n0 + (lane & 15)) * Kp  + bkb;
  v8f acc = {};
  for (int k0 = 0; k0 < Kp; k0 += 32) {
    v8h alo = *(const v8h*)(ap + k0);
    v8h ahi = *(const v8h*)(ap + k0 + 8);
    v16h a;
#pragma unroll
    for (int j = 0; j < 8; ++j) { a[j] = alo[j]; a[8 + j] = ahi[j]; }
    v16h b = *(const v16h*)(bp + k0);
    __builtin_prefetch(ap + k0 + 32, 0, 1);
    __builtin_prefetch(bp + k0 + 32, 0, 1);
    acc = __builtin_amdgcn_wmma_f32_16x16x32_f16(false, a, false, b, (short)0, acc,
                                                 false, false);
  }
  int nc = n0 + (lane & 15);
  float bv = bias ? bias[nc] : 0.0f;
  int mb = m0 + (hi ? 8 : 0);
#pragma unroll
  for (int rr = 0; rr < 8; ++rr) {       // D: VGPR r -> row m0+(hi?8:0)+r, col n0+(lane&15)
    float v = acc[rr] + bv;
    if (relu) v = fmaxf(v, 0.0f);
    int mr = mb + rr;
    if (mr < Mact && nc < Nact) {
      if (C32) C32[(size_t)mr * ldc + nc] = v;
      else     C16[(size_t)mr * ldc + nc] = (h16)v;
    }
  }
}

// ---------------------------------------------------------------- attention core
__global__ void k_attn_core(const h16* __restrict__ Q, const h16* __restrict__ Kb,
                            const h16* __restrict__ Vb, const int* __restrict__ nb,
                            int seg0, int Mact,
                            h16* __restrict__ outb, int* __restrict__ flags) {
  __shared__ float sc[2][KNB];
  __shared__ float attn[2][KNB];
  __shared__ int   msk[KNB];
  int i = blockIdx.x;
  int t = threadIdx.x;
  if (i >= Mact) {
    for (int c = t; c < QP; c += blockDim.x) outb[(size_t)i * QP + c] = (h16)0.0f;
    if (t == 0) flags[i] = 1;
    return;
  }
  if (t < 2 * KNB) {
    int hh = t / KNB, kk = t % KNB;
    int nid = nb[(seg0 + i) * KNB + kk];
    if (hh == 0) msk[kk] = (nid == 0);
    const h16* qp = Q  + (size_t)i * QP + hh * DD;
    const h16* kp = Kb + (size_t)(i * KNB + kk) * QP + hh * DD;
    float s = 0.0f;
    for (int d = 0; d < DD; ++d) s += (float)qp[d] * (float)kp[d];
    s *= 0.07624928516630233f;                     // 1/sqrt(172)
    sc[hh][kk] = (nid == 0) ? -1e9f : s;
  }
  __syncthreads();
  if (t < 2) {
    float mx = -1e30f;
    for (int kk = 0; kk < KNB; ++kk) mx = fmaxf(mx, sc[t][kk]);
    float sum = 0.0f;
    for (int kk = 0; kk < KNB; ++kk) { float e = expf(sc[t][kk] - mx); attn[t][kk] = e; sum += e; }
    float inv = 1.0f / sum;
    for (int kk = 0; kk < KNB; ++kk) attn[t][kk] *= inv;
  }
  if (t == 0) {
    int all = 1;
    for (int kk = 0; kk < KNB; ++kk) all &= msk[kk];
    flags[i] = all;
  }
  __syncthreads();
  for (int c = t; c < QP; c += blockDim.x) {
    float acc = 0.0f;
    if (c < QDIM) {
      int hh = c / DD;
      for (int kk = 0; kk < KNB; ++kk)
        acc += attn[hh][kk] * (float)Vb[(size_t)(i * KNB + kk) * QP + c];
    }
    outb[(size_t)i * QP + c] = (h16)acc;
  }
}

// merge input: [ masked(out@Wo+bo) | src_conv | 0-pad ]  (544 cols)
__global__ void k_build_mrg(const h16* __restrict__ outp, const int* __restrict__ flags,
                            const h16* __restrict__ srcb,
                            h16* __restrict__ mrg, int Mact, int Mpad) {
  int g = blockIdx.x * blockDim.x + threadIdx.x;
  if (g >= Mpad * KP) return;
  int i = g / KP, c = g % KP;
  float v = 0.0f;
  if (i < Mact) {
    if (c < QDIM)      v = flags[i] ? 0.0f : (float)outp[(size_t)i * QP + c];
    else if (c < KDIM) v = (float)srcb[(size_t)i * DP + (c - QDIM)];
  }
  mrg[g] = (h16)v;
}

// ================================================================ host
extern "C" void kernel_launch(void* const* d_in, const int* in_sizes, int n_in,
                              void* d_out, int out_size, void* d_ws, size_t ws_size,
                              hipStream_t stream) {
  (void)in_sizes; (void)n_in; (void)out_size; (void)ws_size;
  const float* nodef   = (const float*)d_in[0];
  const float* edgef   = (const float*)d_in[1];
  const float* memf    = (const float*)d_in[2];
  const float* last_up = (const float*)d_in[3];
  const float* time_w  = (const float*)d_in[4];
  const float* time_b  = (const float*)d_in[5];
  const float* num_w1  = (const float*)d_in[6];
  const float* num_b1  = (const float*)d_in[7];
  const float* num_w2  = (const float*)d_in[8];
  const float* num_b2  = (const float*)d_in[9];
  const float* msg_w1  = (const float*)d_in[10];
  const float* msg_b1  = (const float*)d_in[11];
  const float* msg_w2  = (const float*)d_in[12];
  const float* msg_b2  = (const float*)d_in[13];
  const float* gru_wih = (const float*)d_in[14];
  const float* gru_whh = (const float*)d_in[15];
  const float* gru_bih = (const float*)d_in[16];
  const float* gru_bhh = (const float*)d_in[17];
  const float* att_wq  = (const float*)d_in[18];
  const float* att_wk  = (const float*)d_in[19];
  const float* att_wv  = (const float*)d_in[20];
  const float* att_bq  = (const float*)d_in[21];
  const float* att_bk  = (const float*)d_in[22];
  const float* att_bv  = (const float*)d_in[23];
  const float* att_wo  = (const float*)d_in[24];
  const float* att_bo  = (const float*)d_in[25];
  const float* mrg_w1  = (const float*)d_in[26];
  const float* mrg_b1  = (const float*)d_in[27];
  const float* mrg_w2  = (const float*)d_in[28];
  const float* mrg_b2  = (const float*)d_in[29];
  const float* edge_times = (const float*)d_in[30];
  const float* num_conn   = (const float*)d_in[31];
  const float* n1_et   = (const float*)d_in[32];
  const float* n2_et   = (const float*)d_in[33];
  const int* src       = (const int*)d_in[34];
  const int* dst       = (const int*)d_in[35];
  const int* neg       = (const int*)d_in[36];
  const int* edge_idxs = (const int*)d_in[37];
  const int* n1_nb     = (const int*)d_in[38];
  const int* n1_eidx   = (const int*)d_in[39];
  const int* n2_nb     = (const int*)d_in[40];
  const int* n2_eidx   = (const int*)d_in[41];

  size_t off = 0;
  auto alloc = [&](size_t bytes) -> void* {
    void* p = (char*)d_ws + off;
    off += (bytes + 255) & ~(size_t)255;
    return p;
  };
  float* upd   = (float*)alloc(600 * DD * 4);
  int*   map   = (int*)  alloc((size_t)NDN * 4);
  int*   nodes = (int*)  alloc(912 * 4);
  h16* wqT = (h16*)alloc((size_t)2 * QP * QP * 2);
  h16* wkT = (h16*)alloc((size_t)2 * QP * KP * 2);
  h16* wvT = (h16*)alloc((size_t)2 * QP * KP * 2);
  h16* woT = (h16*)alloc((size_t)2 * QP * QP * 2);
  h16* w1T = (h16*)alloc((size_t)2 * DP * KP * 2);
  h16* w2T = (h16*)alloc((size_t)2 * DP * DP * 2);
  float* bqP = (float*)alloc(2 * QP * 4);
  float* bkP = (float*)alloc(2 * QP * 4);
  float* bvP = (float*)alloc(2 * QP * 4);
  float* boP = (float*)alloc(2 * QP * 4);
  float* b1P = (float*)alloc(2 * DP * 4);
  float* b2P = (float*)alloc(2 * DP * 4);
  h16* featNodes = (h16*)alloc((size_t)912  * DP * 2);
  h16* featN1    = (h16*)alloc((size_t)9008 * DP * 2);
  h16* embA      = (h16*)alloc((size_t)912  * DP * 2);
  h16* neigh1    = (h16*)alloc((size_t)9008 * DP * 2);
  h16* qin  = (h16*)alloc((size_t)1504  * QP * 2);
  h16* kin  = (h16*)alloc((size_t)15040 * KP * 2);
  h16* Qb   = (h16*)alloc((size_t)1504  * QP * 2);
  h16* Kb   = (h16*)alloc((size_t)15040 * QP * 2);
  h16* Vb   = (h16*)alloc((size_t)15040 * QP * 2);
  h16* attb = (h16*)alloc((size_t)1504  * QP * 2);
  h16* outp = (h16*)alloc((size_t)1504  * QP * 2);
  h16* mrgb = (h16*)alloc((size_t)1504  * KP * 2);
  h16* hh1  = (h16*)alloc((size_t)1504  * DP * 2);
  int* flags = (int*)alloc(1504 * 4);

  const int TH = 256;
  // memory delta
  k_init_map<<<(NDN + TH - 1) / TH, TH, 0, stream>>>(map);
  k_scatter_map<<<3, TH, 0, stream>>>(src, dst, map);
  k_msggru<<<600, TH, 0, stream>>>(memf, edgef, last_up, time_w, time_b,
                                   num_w1, num_b1, num_w2, num_b2,
                                   msg_w1, msg_b1, msg_w2, msg_b2,
                                   gru_wih, gru_whh, gru_bih, gru_bhh,
                                   num_conn, edge_times, src, dst, edge_idxs, upd);
  // weights -> f16 transposed + padded
  for (int l = 0; l < 2; ++l) {
    auto CW = [&](const float* W, int Kd, int Nd, h16* Bt, int Kp, int Np) {
      int tot = Kp * Np;
      k_conv_weight<<<(tot + TH - 1) / TH, TH, 0, stream>>>(W, Kd, Nd, Bt, Kp, Np);
    };
    CW(att_wq + (size_t)l * QDIM * QDIM, QDIM, QDIM, wqT + (size_t)l * QP * QP, QP, QP);
    CW(att_wk + (size_t)l * KDIM * QDIM, KDIM, QDIM, wkT + (size_t)l * QP * KP, KP, QP);
    CW(att_wv + (size_t)l * KDIM * QDIM, KDIM, QDIM, wvT + (size_t)l * QP * KP, KP, QP);
    CW(att_wo + (size_t)l * QDIM * QDIM, QDIM, QDIM, woT + (size_t)l * QP * QP, QP, QP);
    CW(mrg_w1 + (size_t)l * KDIM * DD,   KDIM, DD,   w1T + (size_t)l * DP * KP, KP, DP);
    CW(mrg_w2 + (size_t)l * DD * DD,     DD,   DD,   w2T + (size_t)l * DP * DP, DP, DP);
    k_pad_bias<<<2, TH, 0, stream>>>(att_bq + l * QDIM, QDIM, bqP + l * QP, QP);
    k_pad_bias<<<2, TH, 0, stream>>>(att_bk + l * QDIM, QDIM, bkP + l * QP, QP);
    k_pad_bias<<<2, TH, 0, stream>>>(att_bv + l * QDIM, QDIM, bvP + l * QP, QP);
    k_pad_bias<<<2, TH, 0, stream>>>(att_bo + l * QDIM, QDIM, boP + l * QP, QP);
    k_pad_bias<<<1, TH, 0, stream>>>(mrg_b1 + l * DD, DD, b1P + l * DP, DP);
    k_pad_bias<<<1, TH, 0, stream>>>(mrg_b2 + l * DD, DD, b2P + l * DP, DP);
  }
  // layer-0 feature gathers
  k_nodes<<<4, TH, 0, stream>>>(src, dst, neg, nodes);
  k_gather_feat<<<(912 * DP + TH - 1) / TH, TH, 0, stream>>>(nodes, 900, 912, map, upd,
                                                             memf, nodef, featNodes);
  k_gather_feat<<<(9008 * DP + TH - 1) / TH, TH, 0, stream>>>(n1_nb, 9000, 9008, map, upd,
                                                              memf, nodef, featN1);

  auto run_att = [&](int l, int M, int Mpad, const h16* srcb,
                     const int* nbp, const int* eip, const float* etp,
                     int rep, int seg0, int mode, const h16* neighb,
                     h16* out16, float* out32, int ldco, int MactOut, int NactOut) {
    int pairs = M * KNB;
    int Ppad = (pairs + 15) & ~15;
    k_build_qin<<<(Mpad * QP + TH - 1) / TH, TH, 0, stream>>>(srcb, time_b, qin, M, Mpad);
    k_build_kin<<<(Ppad * KP + TH - 1) / TH, TH, 0, stream>>>(mode, nbp, eip, etp, neighb,
        map, upd, memf, nodef, edgef, edge_times, time_w, time_b, rep, seg0, M, kin, Ppad);
    dim3 gQ(Mpad / 16, 3), gKV(Ppad / 16, 3), gD(Mpad / 16, 2);
    k_gemm<<<gQ, TH, 0, stream>>>(qin, QP, wqT + (size_t)l * QP * QP, QP, bqP + l * QP, 0,
                                  Qb, nullptr, QP, QP, Mpad, QP);
    k_gemm<<<gKV, TH, 0, stream>>>(kin, KP, wkT + (size_t)l * QP * KP, KP, bkP + l * QP, 0,
                                   Kb, nullptr, QP, QP, Ppad, QP);
    k_gemm<<<gKV, TH, 0, stream>>>(kin, KP, wvT + (size_t)l * QP * KP, KP, bvP + l * QP, 0,
                                   Vb, nullptr, QP, QP, Ppad, QP);
    k_attn_core<<<Mpad, 64, 0, stream>>>(Qb, Kb, Vb, nbp, seg0, M, attb, flags);
    k_gemm<<<gQ, TH, 0, stream>>>(attb, QP, woT + (size_t)l * QP * QP, QP, boP + l * QP, 0,
                                  outp, nullptr, QP, QP, Mpad, QP);
    k_build_mrg<<<(Mpad * KP + TH - 1) / TH, TH, 0, stream>>>(outp, flags, srcb, mrgb, M, Mpad);
    k_gemm<<<gD, TH, 0, stream>>>(mrgb, KP, w1T + (size_t)l * DP * KP, KP, b1P + l * DP, 1,
                                  hh1, nullptr, DP, DP, Mpad, DP);
    k_gemm<<<gD, TH, 0, stream>>>(hh1, DP, w2T + (size_t)l * DP * DP, DP, b2P + l * DP, 0,
                                  out16, out32, ldco, DP, MactOut, NactOut);
  };

  // call A: layer-0 embedding of target nodes -> embA (f16)
  run_att(0, 900, 912, featNodes, n1_nb, n1_eidx, n1_et, 1, 0, 0, nullptr,
          embA, nullptr, DP, 900, DP);
  // call B: layer-0 embedding of first-hop neighbors -> neigh1 (f16), 6 segments
  for (int s = 0; s < 6; ++s)
    run_att(0, 1500, 1504, featN1 + (size_t)s * 1500 * DP, n2_nb, n2_eidx, n2_et,
            KNB, s * 1500, 0, nullptr, neigh1 + (size_t)s * 1500 * DP, nullptr, DP, 1500, DP);
  // call C: layer-1 aggregation -> d_out (f32 [900,172])
  run_att(1, 900, 912, embA, n1_nb, n1_eidx, n1_et, 1, 0, 1, neigh1,
          nullptr, (float*)d_out, DD, 900, DD);
}
